// Block_21715354649710
// MI455X (gfx1250) — compile-verified, matrix operand order checked by reading, not run on hardware
//
#include <hip/hip_runtime.h>
#include <hip/hip_bf16.h>
#include <math.h>

typedef _Float16 f16;
typedef __attribute__((ext_vector_type(16))) _Float16 v16h;
typedef __attribute__((ext_vector_type(8)))  float    v8f;

#define BDIM 8
#define NDIM 1024
#define CDIM 1024
#define HDIM 16
#define DDIM 64
#define HID  4096
#define ROWS (BDIM * NDIM)   // 8192
#define EPSV 1e-5f

// ---------------------------------------------------------------------------
// CDNA5 async global->LDS copy (16B per lane), tracked by ASYNCcnt.
// ldsoff = low 32 bits of flat address of the LDS destination (== LDS offset).
// ---------------------------------------------------------------------------
__device__ __forceinline__ void async_ld_b128(unsigned ldsoff, const void* gptr) {
  asm volatile("global_load_async_to_lds_b128 %0, %1, off"
               :: "v"(ldsoff), "v"((unsigned long long)(uintptr_t)gptr)
               : "memory");
}
__device__ __forceinline__ void wait_async0() {
  asm volatile("s_wait_asynccnt 0x0" ::: "memory");
}
__device__ __forceinline__ unsigned lds_off(const void* p) {
  return (unsigned)(uintptr_t)p;   // flat LDS addr low 32 bits == LDS offset
}

// ---------------------------------------------------------------------------
// f32 -> f16 conversion (grid-stride)
// ---------------------------------------------------------------------------
__global__ __launch_bounds__(256) void cvt_f32_to_f16(const float* __restrict__ in,
                                                      f16* __restrict__ out, size_t n) {
  size_t i = (size_t)blockIdx.x * blockDim.x + threadIdx.x;
  size_t stride = (size_t)gridDim.x * blockDim.x;
  for (; i < n; i += stride) out[i] = (f16)in[i];
}

// ---------------------------------------------------------------------------
// LayerNorm over C=1024, one block (256 threads) per row, f16 output
// ---------------------------------------------------------------------------
__global__ __launch_bounds__(256) void ln_kernel(const float* __restrict__ x,
                                                 const float* __restrict__ g,
                                                 const float* __restrict__ b,
                                                 f16* __restrict__ out) {
  __shared__ float red[8];
  const int row = blockIdx.x;
  const int t = threadIdx.x;
  const float* xr = x + (size_t)row * CDIM;

  float v[4];
  float s = 0.f;
#pragma unroll
  for (int i = 0; i < 4; ++i) { v[i] = xr[t + i * 256]; s += v[i]; }
#pragma unroll
  for (int m = 16; m >= 1; m >>= 1) s += __shfl_xor(s, m, 32);
  if ((t & 31) == 0) red[t >> 5] = s;
  __syncthreads();
  float tot = 0.f;
#pragma unroll
  for (int w = 0; w < 8; ++w) tot += red[w];
  const float mean = tot * (1.f / CDIM);
  __syncthreads();

  float s2 = 0.f;
#pragma unroll
  for (int i = 0; i < 4; ++i) { float d = v[i] - mean; s2 += d * d; }
#pragma unroll
  for (int m = 16; m >= 1; m >>= 1) s2 += __shfl_xor(s2, m, 32);
  if ((t & 31) == 0) red[t >> 5] = s2;
  __syncthreads();
  float tot2 = 0.f;
#pragma unroll
  for (int w = 0; w < 8; ++w) tot2 += red[w];
  const float rstd = rsqrtf(tot2 * (1.f / CDIM) + EPSV);

#pragma unroll
  for (int i = 0; i < 4; ++i) {
    int c = t + i * 256;
    out[(size_t)row * CDIM + c] = (f16)((v[i] - mean) * rstd * g[c] + b[c]);
  }
}

// ---------------------------------------------------------------------------
// WMMA GEMM: Y = epilogue(A @ W^T + bias)
//   A [M,K] f16 row-major, W [Nout,K] f16 row-major.
//   Block tile 128x128, K-step 32, DOUBLE-BUFFERED LDS with async copies.
//   8 waves: wave (wm,wn) in 2x4 grid, each wave owns 64x32 = 4x2 WMMA tiles.
//   EPI: 0 = f32 out, 1 = f16 out, 2 = f16 GELU out
// ---------------------------------------------------------------------------
template <int EPI>
__global__ __launch_bounds__(256) void gemm_f16_kernel(const f16* __restrict__ A,
                                                       const f16* __restrict__ W,
                                                       const float* __restrict__ bias,
                                                       void* __restrict__ Y,
                                                       int M, int Nout, int K) {
  // row stride 48 f16 = 96 bytes -> every 16-element fragment slice 32B aligned
  __shared__ f16 sA[2][128][48];
  __shared__ f16 sB[2][128][48];

  const int t = threadIdx.x;
  const int wave = t >> 5, lane = t & 31;
  const int lr = lane & 15, hi = lane >> 4;
  const int wm = wave >> 2;   // 0..1
  const int wn = wave & 3;    // 0..3
  const int m0 = blockIdx.y * 128;
  const int n0 = blockIdx.x * 128;

  v8f acc[4][2] = {};

  const int ldr = t >> 1;            // 0..127 tile row for global->LDS copy
  const int ldc = (t & 1) * 16;      // 0 or 16 (f16 elements)

  // Per-thread 32B chunk of each tile = 2 async b128 per array (4 per step).
  auto issue_tiles = [&](int buf, int k0) {
    const f16* ga = &A[(size_t)(m0 + ldr) * K + k0 + ldc];
    const f16* gb = &W[(size_t)(n0 + ldr) * K + k0 + ldc];
    unsigned la = lds_off(&sA[buf][ldr][ldc]);
    unsigned lb = lds_off(&sB[buf][ldr][ldc]);
    async_ld_b128(la,      ga);
    async_ld_b128(la + 16, (const char*)ga + 16);
    async_ld_b128(lb,      gb);
    async_ld_b128(lb + 16, (const char*)gb + 16);
  };

  const int nk = K >> 5;
  issue_tiles(0, 0);

  for (int kt = 0; kt < nk; ++kt) {
    const int cur = kt & 1;
    wait_async0();        // my tile-(kt) copies have landed in LDS
    __syncthreads();      // everyone's copies landed; prev compute done
    if (kt + 1 < nk) issue_tiles(cur ^ 1, (kt + 1) << 5);  // overlap with WMMA

    v16h bfrag[2];
#pragma unroll
    for (int j = 0; j < 2; ++j)
      bfrag[j] = *(const v16h*)&sB[cur][wn * 32 + j * 16 + lr][hi * 16];
#pragma unroll
    for (int i = 0; i < 4; ++i) {
      v16h afrag = *(const v16h*)&sA[cur][wm * 64 + i * 16 + lr][hi * 16];
#pragma unroll
      for (int j = 0; j < 2; ++j)
        acc[i][j] = __builtin_amdgcn_wmma_f32_16x16x32_f16(
            false, afrag, false, bfrag[j], (short)0, acc[i][j], false, false);
    }
  }

  // C/D layout: VGPR v -> row = v + 8*hi within tile, col = lane&15
#pragma unroll
  for (int i = 0; i < 4; ++i) {
#pragma unroll
    for (int j = 0; j < 2; ++j) {
      const int gn = n0 + wn * 32 + j * 16 + lr;
      const float bv = bias ? bias[gn] : 0.f;
      const int gmb = m0 + wm * 64 + i * 16 + 8 * hi;
#pragma unroll
      for (int v = 0; v < 8; ++v) {
        float val = acc[i][j][v] + bv;
        if (EPI == 2) val = 0.5f * val * (1.f + erff(val * 0.70710678118654752f));
        const size_t idx = (size_t)(gmb + v) * Nout + gn;
        if (EPI == 0) ((float*)Y)[idx] = val;
        else          ((f16*)Y)[idx] = (f16)val;
      }
    }
  }
}

// ---------------------------------------------------------------------------
// Flash attention. qkv f16 [B,N,3,H,D] flattened; out f16 [B,N,C].
// Grid: (N/128, H, B); 256 threads = 8 waves. Wave w owns query rows
// [16w,16w+16) of the 128-query block; key blocks of 128 are streamed.
// K tile comes in via async-to-LDS; V is transposed through VGPRs.
// ---------------------------------------------------------------------------
__global__ __launch_bounds__(256) void flash_attn_kernel(const f16* __restrict__ qkv,
                                                         f16* __restrict__ out) {
  const int qb = blockIdx.x, h = blockIdx.y, b = blockIdx.z;
  const int t = threadIdx.x;
  const int wave = t >> 5, lane = t & 31;
  const int lr = lane & 15, hi = lane >> 4;

  __shared__ f16 sQ[128][80];     // 160B rows (32B-aligned slices)
  __shared__ f16 sK[128][80];
  __shared__ f16 sVt[64][144];    // V transposed: [d][key]
  __shared__ f16 sP[128][144];    // probabilities

  const size_t base = (size_t)b * NDIM * (3 * CDIM);
  const int q0 = qb * 128;
  const int r = t >> 1;           // 0..127
  const int d0 = (t & 1) * 32;    // 0 or 32

  // Load Q block, pre-scaled by D^-1/2 = 0.125
  {
    const f16* src = qkv + base + (size_t)(q0 + r) * (3 * CDIM) + h * DDIM + d0;
#pragma unroll
    for (int i = 0; i < 32; ++i) sQ[r][d0 + i] = (f16)((float)src[i] * 0.125f);
  }

  v8f accO[4] = {};
  float m_run[8], l_run[8];
#pragma unroll
  for (int v = 0; v < 8; ++v) { m_run[v] = -1e30f; l_run[v] = 0.f; }

  for (int kb = 0; kb < NDIM / 128; ++kb) {
    __syncthreads();   // prev iteration's P@V done -> safe to refill sK/sVt
    {
      const f16* kp = qkv + base + (size_t)(kb * 128 + r) * (3 * CDIM) + CDIM + h * DDIM + d0;
      const f16* vp = qkv + base + (size_t)(kb * 128 + r) * (3 * CDIM) + 2 * CDIM + h * DDIM + d0;
      unsigned lk = lds_off(&sK[r][d0]);
      async_ld_b128(lk,      kp);                      // K tile: async path
      async_ld_b128(lk + 16, (const char*)kp + 16);
#pragma unroll
      for (int i = 0; i < 32; ++i) sVt[d0 + i][r] = vp[i];   // V: transpose on store
    }
    wait_async0();
    __syncthreads();

    // S = Q @ K^T  (wave's 16 query rows x 128 key cols)
    v8f accS[8] = {};
#pragma unroll
    for (int ks = 0; ks < 2; ++ks) {
      v16h aq = *(const v16h*)&sQ[wave * 16 + lr][ks * 32 + hi * 16];
#pragma unroll
      for (int j = 0; j < 8; ++j) {
        v16h bk = *(const v16h*)&sK[j * 16 + lr][ks * 32 + hi * 16];
        accS[j] = __builtin_amdgcn_wmma_f32_16x16x32_f16(
            false, aq, false, bk, (short)0, accS[j], false, false);
      }
    }

    // Online softmax in accumulator layout: lane (v,hi) holds row v+8*hi,
    // col = 16j + (lane&15). Reduce across the 16-lane half (xor 8,4,2,1).
#pragma unroll
    for (int v = 0; v < 8; ++v) {
      float bm = -1e30f;
#pragma unroll
      for (int j = 0; j < 8; ++j) bm = fmaxf(bm, accS[j][v]);
#pragma unroll
      for (int m = 8; m >= 1; m >>= 1) bm = fmaxf(bm, __shfl_xor(bm, m, 32));
      const float mn = fmaxf(m_run[v], bm);
      float rs = 0.f;
#pragma unroll
      for (int j = 0; j < 8; ++j) {
        float e = __expf(accS[j][v] - mn);
        rs += e;
        sP[wave * 16 + v + 8 * hi][j * 16 + lr] = (f16)e;
      }
#pragma unroll
      for (int m = 8; m >= 1; m >>= 1) rs += __shfl_xor(rs, m, 32);
      const float a = __expf(m_run[v] - mn);
      l_run[v] = l_run[v] * a + rs;
      m_run[v] = mn;
#pragma unroll
      for (int j2 = 0; j2 < 4; ++j2) accO[j2][v] *= a;
    }
    __syncthreads();

    // O += P @ V   (k-dim = key index, contiguous in sP rows / sVt rows)
#pragma unroll
    for (int ks2 = 0; ks2 < 4; ++ks2) {
      v16h ap = *(const v16h*)&sP[wave * 16 + lr][ks2 * 32 + hi * 16];
#pragma unroll
      for (int j2 = 0; j2 < 4; ++j2) {
        v16h bv = *(const v16h*)&sVt[j2 * 16 + lr][ks2 * 32 + hi * 16];
        accO[j2] = __builtin_amdgcn_wmma_f32_16x16x32_f16(
            false, ap, false, bv, (short)0, accO[j2], false, false);
      }
    }
  }

  // Normalize and store
#pragma unroll
  for (int j2 = 0; j2 < 4; ++j2) {
#pragma unroll
    for (int v = 0; v < 8; ++v) {
      const int row = q0 + wave * 16 + v + 8 * hi;
      const float o = accO[j2][v] / l_run[v];
      out[((size_t)b * NDIM + row) * CDIM + h * DDIM + j2 * 16 + lr] = (f16)o;
    }
  }
}

// ---------------------------------------------------------------------------
// out = xin + gamma[c] * y   (elementwise, C=1024 power of two)
// ---------------------------------------------------------------------------
__global__ __launch_bounds__(256) void residual_ls_kernel(const float* __restrict__ xin,
                                                          const float* __restrict__ y,
                                                          const float* __restrict__ gamma,
                                                          float* __restrict__ out, size_t n) {
  size_t i = (size_t)blockIdx.x * blockDim.x + threadIdx.x;
  size_t stride = (size_t)gridDim.x * blockDim.x;
  for (; i < n; i += stride)
    out[i] = xin[i] + gamma[i & (CDIM - 1)] * y[i];
}

// ---------------------------------------------------------------------------
extern "C" void kernel_launch(void* const* d_in, const int* in_sizes, int n_in,
                              void* d_out, int out_size, void* d_ws, size_t ws_size,
                              hipStream_t stream) {
  (void)in_sizes; (void)n_in; (void)out_size; (void)ws_size;
  const float* x        = (const float*)d_in[0];
  const float* ln1_g    = (const float*)d_in[1];
  const float* ln1_b    = (const float*)d_in[2];
  const float* w_qkv    = (const float*)d_in[3];
  const float* w_proj   = (const float*)d_in[4];
  const float* b_proj   = (const float*)d_in[5];
  const float* ls1      = (const float*)d_in[6];
  const float* ln2_g    = (const float*)d_in[7];
  const float* ln2_b    = (const float*)d_in[8];
  const float* w_fc1    = (const float*)d_in[9];
  const float* b_fc1    = (const float*)d_in[10];
  const float* w_fc2    = (const float*)d_in[11];
  const float* b_fc2    = (const float*)d_in[12];
  const float* ls2      = (const float*)d_in[13];

  char* w = (char*)d_ws;
  f16* wqkv16  = (f16*)w; w += (size_t)3 * CDIM * CDIM * 2;
  f16* wproj16 = (f16*)w; w += (size_t)CDIM * CDIM * 2;
  f16* wfc1_16 = (f16*)w; w += (size_t)HID * CDIM * 2;
  f16* wfc2_16 = (f16*)w; w += (size_t)CDIM * HID * 2;
  f16* xn16    = (f16*)w; w += (size_t)ROWS * CDIM * 2;
  f16* qkv16   = (f16*)w; w += (size_t)ROWS * 3 * CDIM * 2;
  f16* attn16  = (f16*)w; w += (size_t)ROWS * CDIM * 2;
  f16* h16     = (f16*)w; w += (size_t)ROWS * HID * 2;
  float* yf32  = (float*)w; w += (size_t)ROWS * CDIM * 4;
  float* x1    = (float*)w; w += (size_t)ROWS * CDIM * 4;

  // 1. Convert weights to f16
  cvt_f32_to_f16<<<2048, 256, 0, stream>>>(w_qkv,  wqkv16,  (size_t)3 * CDIM * CDIM);
  cvt_f32_to_f16<<<2048, 256, 0, stream>>>(w_proj, wproj16, (size_t)CDIM * CDIM);
  cvt_f32_to_f16<<<2048, 256, 0, stream>>>(w_fc1,  wfc1_16, (size_t)HID * CDIM);
  cvt_f32_to_f16<<<2048, 256, 0, stream>>>(w_fc2,  wfc2_16, (size_t)CDIM * HID);

  // 2. LN1
  ln_kernel<<<ROWS, 256, 0, stream>>>(x, ln1_g, ln1_b, xn16);

  // 3. QKV projection: [8192,3072] = xn16 @ wqkv^T  (f16 out, no bias)
  gemm_f16_kernel<1><<<dim3(3 * CDIM / 128, ROWS / 128), 256, 0, stream>>>(
      xn16, wqkv16, nullptr, qkv16, ROWS, 3 * CDIM, CDIM);

  // 4. Flash attention
  flash_attn_kernel<<<dim3(NDIM / 128, HDIM, BDIM), 256, 0, stream>>>(qkv16, attn16);

  // 5. Output projection (f32 out, bias)
  gemm_f16_kernel<0><<<dim3(CDIM / 128, ROWS / 128), 256, 0, stream>>>(
      attn16, wproj16, b_proj, yf32, ROWS, CDIM, CDIM);

  // 6. x1 = x + ls1 * attn
  residual_ls_kernel<<<8192, 256, 0, stream>>>(x, yf32, ls1, x1, (size_t)ROWS * CDIM);

  // 7. LN2
  ln_kernel<<<ROWS, 256, 0, stream>>>(x1, ln2_g, ln2_b, xn16);

  // 8. FC1 + GELU (f16 out)
  gemm_f16_kernel<2><<<dim3(HID / 128, ROWS / 128), 256, 0, stream>>>(
      xn16, wfc1_16, b_fc1, h16, ROWS, HID, CDIM);

  // 9. FC2 (f32 out)
  gemm_f16_kernel<0><<<dim3(CDIM / 128, ROWS / 128), 256, 0, stream>>>(
      h16, wfc2_16, b_fc2, yf32, ROWS, CDIM, HID);

  // 10. out = x1 + ls2 * mlp
  residual_ls_kernel<<<8192, 256, 0, stream>>>(x1, yf32, ls2, (float*)d_out,
                                               (size_t)ROWS * CDIM);
}
